// up_block_47845935677649
// MI455X (gfx1250) — matrix operand order repaired and checked by reading, not compiled, not run on hardware
//
#include <hip/hip_runtime.h>
#include <hip/hip_bf16.h>

// ---------------------------------------------------------------------------
// up_block for MI455X (gfx1250, wave32, WMMA bf16 -> f32 accumulate)
// Workgroup tile: 128 rows x 64 cols; 4 waves; each wave: 2 M-tiles x 4 N-tiles
// so every B fragment loaded from LDS is reused by two WMMAs.
// ---------------------------------------------------------------------------

#define N_RAW 40962
#define N_NEW 163842            // 4*N_RAW - 6
#define IN_CH 128
#define OUT_CH 64
#define NEG_SLOPE 0.2f
#define BN_EPS 1e-5f

typedef __bf16 bf16x16 __attribute__((ext_vector_type(16)));
typedef float  f32x8   __attribute__((ext_vector_type(8)));

union FragU {
    uint4   q[2];   // two 16B ds_load_b128 chunks
    bf16x16 v;      // 16 bf16 values = one WMMA operand per lane
};

__device__ __forceinline__ unsigned short f2bf(float f) {
    unsigned int u = __float_as_uint(f);
    unsigned int r = u + 0x7FFFu + ((u >> 16) & 1u);   // round-to-nearest-even
    return (unsigned short)(r >> 16);
}

// ---------------------------------------------------------------------------
// Kernel 1: upconv GEMM  up[40962,448] = x1[40962,128] @ W_up[128,448] + b_up
// grid: (ceil(N_RAW/128), 7 col-tiles), block: 128 (4 waves)
// ---------------------------------------------------------------------------
__global__ __launch_bounds__(128)
void upconv_gemm_kernel(const float* __restrict__ x1,
                        const float* __restrict__ W_up,
                        const float* __restrict__ b_up,
                        float* __restrict__ up) {
    constexpr int CIN = 128;
    constexpr int PIT = CIN + 8;                 // halves; 272B rows, 16B aligned
    __shared__ __align__(16) unsigned short At[128][PIT];
    __shared__ __align__(16) unsigned short Wt[64][PIT];

    const int tid  = threadIdx.x;
    const int v0   = blockIdx.x * 128;
    const int c0   = blockIdx.y * 64;
    const int lane = tid & 31;
    const int wv   = tid >> 5;
    const int lm   = lane & 15;
    const int hi   = lane >> 4;

    for (int e = tid; e < 128 * CIN; e += 128) {
        int r = e >> 7, c = e & 127;
        int row = v0 + r; if (row >= N_RAW) row = N_RAW - 1;
        At[r][c] = f2bf(x1[row * CIN + c]);
    }
    for (int e = tid; e < 64 * CIN; e += 128) {
        int n = e & 63, k = e >> 6;
        Wt[n][k] = f2bf(W_up[k * 448 + c0 + n]);
    }
    __syncthreads();

    f32x8 acc[2][4];
#pragma unroll
    for (int mt = 0; mt < 2; ++mt)
#pragma unroll
        for (int nt = 0; nt < 4; ++nt)
            acc[mt][nt] = (f32x8){0.f,0.f,0.f,0.f,0.f,0.f,0.f,0.f};

#pragma unroll
    for (int kc = 0; kc < CIN / 32; ++kc) {
        FragU fa[2];
#pragma unroll
        for (int mt = 0; mt < 2; ++mt) {
            const unsigned short* ar = &At[32 * wv + 16 * mt + lm][kc * 32 + hi * 8];
            fa[mt].q[0] = *(const uint4*)(ar);
            fa[mt].q[1] = *(const uint4*)(ar + 16);
        }
#pragma unroll
        for (int nt = 0; nt < 4; ++nt) {
            FragU fb;
            const unsigned short* br = &Wt[nt * 16 + lm][kc * 32 + hi * 8];
            fb.q[0] = *(const uint4*)(br);
            fb.q[1] = *(const uint4*)(br + 16);
#pragma unroll
            for (int mt = 0; mt < 2; ++mt)
                acc[mt][nt] = __builtin_amdgcn_wmma_f32_16x16x32_bf16(
                    false, fa[mt].v, false, fb.v, (short)0, acc[mt][nt],
                    false, false);
        }
    }

#pragma unroll
    for (int mt = 0; mt < 2; ++mt)
#pragma unroll
        for (int nt = 0; nt < 4; ++nt) {
            int col = c0 + nt * 16 + lm;
            float bv = b_up[col];
#pragma unroll
            for (int r = 0; r < 8; ++r) {
                int v = v0 + 32 * wv + 16 * mt + 8 * hi + r;
                if (v < N_RAW) up[v * 448 + col] = acc[mt][nt][r] + bv;
            }
        }
}

// ---------------------------------------------------------------------------
// Kernel 2: build x = concat([top | mean(down pairs)], x2) -> [N_NEW,128]
// Exact torch .view(-1, 64, 2).mean(2) semantics.
// ---------------------------------------------------------------------------
__global__ void build_x_kernel(const float* __restrict__ up,
                               const float* __restrict__ x2,
                               const int* __restrict__ topIdx,
                               const int* __restrict__ downIdx,
                               float* __restrict__ x) {
    int idx = blockIdx.x * blockDim.x + threadIdx.x;
    if (idx >= N_NEW * 128) return;
    int r = idx >> 7, c = idx & 127;
    float val;
    if (c >= 64) {
        val = x2[r * 64 + (c - 64)];
    } else if (r < N_RAW) {
        val = up[topIdx[r] * 64 + c];
    } else {
        int m    = r - N_RAW;
        int half = c >> 5;                 // c<32 -> gathered row 2m; else 2m+1
        int g    = downIdx[2 * m + half];
        int base = (c * 2) & 63;           // 2c mod 64
        val = 0.5f * (up[g * 64 + base] + up[g * 64 + base + 1]);
    }
    x[idx] = val;
}

// ---------------------------------------------------------------------------
// Kernel 3: fused one-ring gather + GEMM
//   h[v,0:64] = sum_j  x[neigh[7v+j], 0:CIN] @ W[j*CIN:(j+1)*CIN, 0:64]  + b
// grid: ceil(N_NEW/128), block 128 (4 waves). K streamed per neighbor via LDS.
// ---------------------------------------------------------------------------
template <int CIN>
__global__ __launch_bounds__(128)
void conv_gather_gemm_kernel(const float* __restrict__ src,
                             const float* __restrict__ W,
                             const float* __restrict__ bias,
                             const int* __restrict__ neigh,
                             float* __restrict__ out,
                             int nrows) {
    constexpr int PIT = CIN + 8;
    __shared__ __align__(16) unsigned short At[128][PIT];
    __shared__ __align__(16) unsigned short Wt[64][PIT];

    const int tid  = threadIdx.x;
    const int v0   = blockIdx.x * 128;
    const int lane = tid & 31;
    const int wv   = tid >> 5;
    const int lm   = lane & 15;
    const int hi   = lane >> 4;

    f32x8 acc[2][4];
#pragma unroll
    for (int mt = 0; mt < 2; ++mt)
#pragma unroll
        for (int nt = 0; nt < 4; ++nt)
            acc[mt][nt] = (f32x8){0.f,0.f,0.f,0.f,0.f,0.f,0.f,0.f};

    for (int j = 0; j < 7; ++j) {
        __syncthreads();   // previous compute phase done before LDS overwrite
        // stage gathered activation slice (bf16)
        for (int e = tid; e < 128 * CIN; e += 128) {
            int r = e / CIN, c = e % CIN;
            int v = v0 + r;
            int g = (v < nrows) ? neigh[7 * v + j] : 0;
            At[r][c] = f2bf(src[g * CIN + c]);
        }
        // stage weight slice transposed: Wt[n][k] = W[(j*CIN+k)*64 + n]
        for (int e = tid; e < 64 * CIN; e += 128) {
            int n = e & 63, k = e >> 6;
            Wt[n][k] = f2bf(W[(j * CIN + k) * 64 + n]);
        }
        if (j < 6 && tid < CIN)   // warm next weight slice (global_prefetch_b8)
            __builtin_prefetch(&W[((j + 1) * CIN + tid) * 64], 0, 1);
        __syncthreads();

#pragma unroll
        for (int kc = 0; kc < CIN / 32; ++kc) {
            FragU fa[2];
#pragma unroll
            for (int mt = 0; mt < 2; ++mt) {
                const unsigned short* ar =
                    &At[32 * wv + 16 * mt + lm][kc * 32 + hi * 8];
                fa[mt].q[0] = *(const uint4*)(ar);
                fa[mt].q[1] = *(const uint4*)(ar + 16);
            }
#pragma unroll
            for (int nt = 0; nt < 4; ++nt) {
                FragU fb;
                const unsigned short* br = &Wt[nt * 16 + lm][kc * 32 + hi * 8];
                fb.q[0] = *(const uint4*)(br);
                fb.q[1] = *(const uint4*)(br + 16);
#pragma unroll
                for (int mt = 0; mt < 2; ++mt)
                    acc[mt][nt] = __builtin_amdgcn_wmma_f32_16x16x32_bf16(
                        false, fa[mt].v, false, fb.v, (short)0, acc[mt][nt],
                        false, false);
            }
        }
    }

#pragma unroll
    for (int mt = 0; mt < 2; ++mt)
#pragma unroll
        for (int nt = 0; nt < 4; ++nt) {
            int col = nt * 16 + lm;
            float bv = bias[col];
#pragma unroll
            for (int r = 0; r < 8; ++r) {
                int v = v0 + 32 * wv + 16 * mt + 8 * hi + r;
                if (v < nrows) out[v * 64 + col] = acc[mt][nt][r] + bv;
            }
        }
}

// ---------------------------------------------------------------------------
// BatchNorm: deterministic two-stage column reduction + fused apply/LeakyReLU
// ---------------------------------------------------------------------------
__global__ void bn_reduce1_kernel(const float* __restrict__ h,
                                  float* __restrict__ partials, int nrows) {
    __shared__ float sdat[512];
    int t = threadIdx.x, col = t & 63, g = t >> 6;     // 4 row-groups per col
    float s = 0.f, s2 = 0.f;
    for (int r = blockIdx.x * 4 + g; r < nrows; r += gridDim.x * 4) {
        float v = h[r * 64 + col];
        s += v; s2 += v * v;
    }
    sdat[t] = s; sdat[256 + t] = s2;
    __syncthreads();
    if (t < 64) {
        float a = sdat[t] + sdat[t + 64] + sdat[t + 128] + sdat[t + 192];
        float b = sdat[256 + t] + sdat[256 + t + 64] +
                  sdat[256 + t + 128] + sdat[256 + t + 192];
        partials[blockIdx.x * 128 + t]      = a;
        partials[blockIdx.x * 128 + 64 + t] = b;
    }
}

__global__ void bn_reduce2_kernel(const float* __restrict__ partials,
                                  float* __restrict__ stats, int nparts) {
    int t = threadIdx.x;
    if (t >= 128) return;
    float s = 0.f;
    for (int p = 0; p < nparts; ++p) s += partials[p * 128 + t];  // fixed order
    stats[t] = s;
}

__global__ void bn_apply_kernel(const float* __restrict__ src,
                                const float* __restrict__ stats,
                                const float* __restrict__ gamma,
                                const float* __restrict__ beta,
                                float* __restrict__ dst, int nrows) {
    int idx = blockIdx.x * blockDim.x + threadIdx.x;
    if (idx >= nrows * 64) return;
    int c = idx & 63;
    float invN = 1.0f / (float)nrows;
    float m    = stats[c] * invN;
    float var  = stats[64 + c] * invN - m * m;          // biased variance
    float y = (src[idx] - m) * rsqrtf(var + BN_EPS) * gamma[c] + beta[c];
    dst[idx] = (y >= 0.f) ? y : NEG_SLOPE * y;
}

// ---------------------------------------------------------------------------
// Launcher
// ---------------------------------------------------------------------------
extern "C" void kernel_launch(void* const* d_in, const int* in_sizes, int n_in,
                              void* d_out, int out_size, void* d_ws, size_t ws_size,
                              hipStream_t stream) {
    const float* x1    = (const float*)d_in[0];
    const float* x2    = (const float*)d_in[1];
    const float* W_up  = (const float*)d_in[2];
    const float* b_up  = (const float*)d_in[3];
    const float* W_c1  = (const float*)d_in[4];
    const float* b_c1  = (const float*)d_in[5];
    const float* g1    = (const float*)d_in[6];
    const float* be1   = (const float*)d_in[7];
    const float* W_c2  = (const float*)d_in[8];
    const float* b_c2  = (const float*)d_in[9];
    const float* g2    = (const float*)d_in[10];
    const float* be2   = (const float*)d_in[11];
    const int* topIdx  = (const int*)d_in[12];
    const int* downIdx = (const int*)d_in[13];
    const int* neigh   = (const int*)d_in[14];
    float* out = (float*)d_out;
    float* ws  = (float*)d_ws;

    // workspace layout (floats)
    float* up       = ws;                                  // 7*N_RAW*64
    float* xcat     = up + (size_t)N_RAW * 7 * 64;         // N_NEW*128
    float* partials = xcat + (size_t)N_NEW * 128;          // 128*128
    float* stats    = partials + 128 * 128;                // 128
    float* h1       = up;                                  // reuse after build_x

    // 1) upconv GEMM
    upconv_gemm_kernel<<<dim3((N_RAW + 127) / 128, 7), 128, 0, stream>>>(
        x1, W_up, b_up, up);

    // 2) assemble x = [x_up | x2]
    build_x_kernel<<<(N_NEW * 128 + 255) / 256, 256, 0, stream>>>(
        up, x2, topIdx, downIdx, xcat);

    // 3) conv1 (gather-GEMM, K=896) -> h1 (overwrites up region)
    conv_gather_gemm_kernel<128><<<(N_NEW + 127) / 128, 128, 0, stream>>>(
        xcat, W_c1, b_c1, neigh, h1, N_NEW);
    bn_reduce1_kernel<<<128, 256, 0, stream>>>(h1, partials, N_NEW);
    bn_reduce2_kernel<<<1, 128, 0, stream>>>(partials, stats, 128);
    bn_apply_kernel<<<(N_NEW * 64 + 255) / 256, 256, 0, stream>>>(
        h1, stats, g1, be1, h1, N_NEW);

    // 4) conv2 (gather-GEMM, K=448) -> d_out, BN+leaky in place
    conv_gather_gemm_kernel<64><<<(N_NEW + 127) / 128, 128, 0, stream>>>(
        h1, W_c2, b_c2, neigh, out, N_NEW);
    bn_reduce1_kernel<<<128, 256, 0, stream>>>(out, partials, N_NEW);
    bn_reduce2_kernel<<<1, 128, 0, stream>>>(partials, stats, 128);
    bn_apply_kernel<<<(N_NEW * 64 + 255) / 256, 256, 0, stream>>>(
        out, stats, g2, be2, out, N_NEW);
}
